// MultiHeadAttention_8340826489502
// MI455X (gfx1250) — matrix operand order repaired and checked by reading, not compiled, not run on hardware
//
#include <hip/hip_runtime.h>

typedef __attribute__((ext_vector_type(16))) _Float16 v16h;
typedef __attribute__((ext_vector_type(8)))  _Float16 v8h;
typedef __attribute__((ext_vector_type(4)))  _Float16 v4h;
typedef __attribute__((ext_vector_type(8)))  float    v8f;
typedef __attribute__((ext_vector_type(4)))  float    v4f;
typedef __attribute__((ext_vector_type(4)))  unsigned int u32x4;
typedef __attribute__((ext_vector_type(8)))  int      i32x8;
typedef __attribute__((ext_vector_type(4)))  int      i32x4;

#define DM 1024
#define NH 16
#define DK 64
#define BB 4
#define SS 2048

#ifndef __has_builtin
#define __has_builtin(x) 0
#endif

// ---- CDNA5 feature probes (device pass only) ------------------------------
#if defined(__AMDGCN__) && __has_builtin(__builtin_amdgcn_global_load_async_to_lds_b128)
#define HAVE_ASYNC 1
#else
#define HAVE_ASYNC 0
#endif

#if defined(__AMDGCN__) && __has_builtin(__builtin_amdgcn_tensor_load_to_lds) && \
    __has_builtin(__builtin_amdgcn_s_wait_tensorcnt)
#define HAVE_TDM 1
#else
#define HAVE_TDM 0
#endif

#if HAVE_ASYNC
// Param types per hipcc diagnostic: (v4i __device__*, v4i __shared__*, Ii, Ii)
using gvec_t = __attribute__((address_space(1))) i32x4*;
using lvec_t = __attribute__((address_space(3))) i32x4*;
__device__ inline void async_cp16(const _Float16* g, _Float16* l) {
  __builtin_amdgcn_global_load_async_to_lds_b128(
      (gvec_t)(const_cast<_Float16*>(g)), (lvec_t)l, 0, 0);
}
#endif

template <int N>
__device__ inline void wait_async() {
#if defined(__AMDGCN__)
#if __has_builtin(__builtin_amdgcn_s_wait_asynccnt)
  __builtin_amdgcn_s_wait_asynccnt(N);
#else
  asm volatile("s_wait_asynccnt %0" ::"n"(N) : "memory");
#endif
#endif
}

#if HAVE_TDM
__device__ inline unsigned lds_off_u32(const void* p) {
  return (unsigned)(unsigned long long)(__attribute__((address_space(3))) const void*)p;
}
// 2D f16 tile load via Tensor Data Mover.  D# layout per CDNA5 ISA ch.8:
// group0: count/flags | lds_addr | global_addr[56:0] | type=2
// group1: mask/data_size/pad | tensor dims, tile dims, strides (elem units).
// pad_interval code: 2^(c+1) DWORDs between pads; pad_amount code: (c+1) DWORDs.
// 6-arg builtin on this toolchain; groups 2/3 + extra group zero for 2D.
__device__ inline void tdm_load_tile_f16(const _Float16* gsrc, unsigned lds_byte,
                                         int rows, int cols, int gstride,
                                         int pad_int_code, int pad_amt_code,
                                         int pad_en) {
  unsigned long long ga = (unsigned long long)gsrc;
  u32x4 g0;
  g0[0] = 1u;                                             // count=1 (valid)
  g0[1] = lds_byte;                                       // lds_addr
  g0[2] = (unsigned)(ga & 0xffffffffull);                 // global_addr[31:0]
  g0[3] = (unsigned)((ga >> 32) & 0x1ffffffull) | (2u << 30);  // [56:32]|type=2
  i32x8 g1;
  g1[0] = (int)((1u << 16) | ((unsigned)pad_en << 20) |
                ((unsigned)pad_int_code << 22) | ((unsigned)pad_amt_code << 25));
  g1[1] = (int)(((unsigned)cols & 0xffffu) << 16);        // tensor_dim0 lo
  g1[2] = (int)((((unsigned)cols >> 16) & 0xffffu) |
                (((unsigned)rows & 0xffffu) << 16));      // dim0 hi | dim1 lo
  g1[3] = (int)((((unsigned)rows >> 16) & 0xffffu) |
                (((unsigned)cols & 0xffffu) << 16));      // dim1 hi | tile_dim0
  g1[4] = (int)((unsigned)rows & 0xffffu);                // tile_dim1, tile_dim2=0
  g1[5] = gstride;                                        // dim0_stride[31:0]
  g1[6] = 0;
  g1[7] = 0;
  i32x4 gz4 = {0, 0, 0, 0};
  i32x8 gz8 = {0, 0, 0, 0, 0, 0, 0, 0};
  __builtin_amdgcn_tensor_load_to_lds(g0, g1, gz4, gz4, gz8, 0);
}
#endif

// ---------------------------------------------------------------------------
// WMMA fragment loaders (LDS row-major, stride in halves), per ISA 7.12.2.
// ---------------------------------------------------------------------------
__device__ inline v16h lds_load_a(const _Float16* p, int lda, int lane) {
  const _Float16* r = p + (lane & 15) * lda + ((lane >> 4) << 3);
  v8h lo = *(const v8h*)(r);
  v8h hi = *(const v8h*)(r + 16);
  v16h a;
#pragma unroll
  for (int i = 0; i < 8; ++i) { a[i] = lo[i]; a[i + 8] = hi[i]; }
  return a;
}

__device__ inline v16h lds_load_b(const _Float16* p, int ldb, int lane) {
  const _Float16* r = p + (lane & 15) * ldb + ((lane >> 4) << 3);
  v8h lo = *(const v8h*)(r);
  v8h hi = *(const v8h*)(r + 16);
  v16h b;
#pragma unroll
  for (int i = 0; i < 8; ++i) { b[i] = lo[i]; b[i + 8] = hi[i]; }
  return b;
}

__device__ inline v8f wmma_f16(v16h a, v16h b, v8f c) {
  return __builtin_amdgcn_wmma_f32_16x16x32_f16(false, a, false, b,
                                                (short)0, c, false, false);
}

// Stage one 128x32 f16 tile (global row stride DM) into LDS (stride 40).
__device__ inline void stage_tile_128x32(const _Float16* g, int k0,
                                         _Float16* l, int tid) {
#pragma unroll
  for (int i = 0; i < 2; ++i) {
    int idx = tid + i * 256;
    int row = idx >> 2;
    int c8  = (idx & 3) << 3;
#if HAVE_ASYNC
    async_cp16(g + (size_t)row * DM + k0 + c8, l + row * 40 + c8);
#else
    *(v8h*)(l + row * 40 + c8) = *(const v8h*)(g + (size_t)row * DM + k0 + c8);
#endif
  }
}

// Shared 8-wave (2x4) 128x128 WMMA compute step on LDS tiles (stride 40).
__device__ inline void gemm_step_128x128(const _Float16* Ab, const _Float16* Bb,
                                         int wm, int wn, int lane,
                                         v8f acc[4][2]) {
  v16h bf[2];
#pragma unroll
  for (int j = 0; j < 2; ++j)
    bf[j] = lds_load_b(Bb + (wn * 32 + j * 16) * 40, 40, lane);
#pragma unroll
  for (int i = 0; i < 4; ++i) {
    v16h af = lds_load_a(Ab + (wm * 64 + i * 16) * 40, 40, lane);
#pragma unroll
    for (int j = 0; j < 2; ++j) acc[i][j] = wmma_f16(af, bf[j], acc[i][j]);
  }
}

// ---------------------------------------------------------------------------
// Kernel 0: f32 -> f16 elementwise convert (inputs and weights, one time).
// ---------------------------------------------------------------------------
__global__ __launch_bounds__(256) void cvt_f16(const float* __restrict__ src,
                                               _Float16* __restrict__ dst,
                                               int n) {
  int i = (blockIdx.x * 256 + threadIdx.x) * 8;
  if (i < n) {
    v4f a = *(const v4f*)(src + i);
    v4f b = *(const v4f*)(src + i + 4);
    v8h h;
#pragma unroll
    for (int t = 0; t < 4; ++t) { h[t] = (_Float16)a[t]; h[4 + t] = (_Float16)b[t]; }
    *(v8h*)(dst + i) = h;
  }
}

// ---------------------------------------------------------------------------
// Kernel 1: fused QKV projection (f16 in, WMMA, async double-buffered LDS,
// last K-iteration peeled so the hot loop has no wait-selection branch).
// z selects Q/K/V.  q,k -> [B,H,S,DK]; v -> transposed [B,H,DK,S].
// ---------------------------------------------------------------------------
__global__ __launch_bounds__(256) void qkv_gemm(
    const _Float16* __restrict__ xh, const _Float16* __restrict__ wh,
    const float* __restrict__ bq, const float* __restrict__ bk,
    const float* __restrict__ bv,
    _Float16* __restrict__ qh, _Float16* __restrict__ kh,
    _Float16* __restrict__ vht) {
  const int z = blockIdx.z;
  const _Float16* X    = xh + (size_t)z * BB * SS * DM;
  const _Float16* W    = wh + (size_t)z * DM * DM;
  const float*    bias = (z == 0) ? bq : (z == 1) ? bk : bv;

  __shared__ _Float16 As[2 * 128 * 40];
  __shared__ _Float16 Bs[2 * 128 * 40];

  const int tid  = threadIdx.x;
  const int lane = tid & 31;
  const int w    = tid >> 5;
  const int wm   = w & 1;
  const int wn   = w >> 1;
  const int m0   = blockIdx.x * 128;
  const int n0   = blockIdx.y * 128;

  const _Float16* Xa = X + (size_t)m0 * DM;
  const _Float16* Wb = W + (size_t)n0 * DM;

  v8f acc[4][2] = {};

  stage_tile_128x32(Xa, 0, As, tid);
  stage_tile_128x32(Wb, 0, Bs, tid);
  int cur = 0;
  for (int k0 = 0; k0 < DM - 32; k0 += 32) {      // steady state: always pipeline
    stage_tile_128x32(Xa, k0 + 32, As + (cur ^ 1) * 128 * 40, tid);
    stage_tile_128x32(Wb, k0 + 32, Bs + (cur ^ 1) * 128 * 40, tid);
    wait_async<4>();             // retire current tile; next stays in flight
    __syncthreads();
    gemm_step_128x128(As + cur * 128 * 40, Bs + cur * 128 * 40, wm, wn, lane, acc);
    __syncthreads();
    cur ^= 1;
  }
  wait_async<0>();               // peeled final tile
  __syncthreads();
  gemm_step_128x128(As + cur * 128 * 40, Bs + cur * 128 * 40, wm, wn, lane, acc);

#pragma unroll
  for (int i = 0; i < 4; ++i) {
    int mbase = m0 + wm * 64 + i * 16 + ((lane >> 4) << 3);
#pragma unroll
    for (int j = 0; j < 2; ++j) {
      int n  = n0 + wn * 32 + j * 16 + (lane & 15);
      float bvn = bias[n];
      int h = n >> 6, d = n & 63;
#pragma unroll
      for (int r = 0; r < 8; ++r) {
        int mm = mbase + r;
        int b  = mm >> 11;
        int s  = mm & (SS - 1);
        float val = acc[i][j][r] + bvn;
        if (z == 2) {
          vht[(((size_t)b * NH + h) * DK + d) * SS + s] = (_Float16)val;
        } else if (z == 0) {
          qh[(((size_t)b * NH + h) * SS + s) * DK + d] = (_Float16)val;
        } else {
          kh[(((size_t)b * NH + h) * SS + s) * DK + d] = (_Float16)val;
        }
      }
    }
  }
}

// ---------------------------------------------------------------------------
// Kernel 2a: scores = (q . k^T) / 8 per head; K=64 resident in LDS.
// Tiles loaded by the Tensor Data Mover (wave 0 issues two tensor_load_to_lds,
// waits TENSORcnt), else async/manual staging.
// ---------------------------------------------------------------------------
__global__ __launch_bounds__(256) void attn_scores(
    const _Float16* __restrict__ qh, const _Float16* __restrict__ kh,
    float* __restrict__ probs) {
  const int bh = blockIdx.z;
  const int s0 = blockIdx.x * 128;
  const int t0 = blockIdx.y * 128;
  if (t0 > s0 + 127) return;         // fully masked block

  __shared__ _Float16 Aq[128 * 72];
  __shared__ _Float16 Bk[128 * 72];

  const int tid  = threadIdx.x;
  const int lane = tid & 31;
  const int w    = tid >> 5;
  const int wm   = w & 1;
  const int wn   = w >> 1;

  const _Float16* qb = qh + (size_t)bh * SS * DK;
  const _Float16* kb = kh + (size_t)bh * SS * DK;

#if HAVE_TDM
  if (w == 0) {
    // 128x64 f16 tiles; LDS rows padded 64->72 halves (pad codes: 32 DW / 4 DW)
    tdm_load_tile_f16(qb + (size_t)s0 * DK, lds_off_u32(Aq), 128, 64, 64, 4, 3, 1);
    tdm_load_tile_f16(kb + (size_t)t0 * DK, lds_off_u32(Bk), 128, 64, 64, 4, 3, 1);
    __builtin_amdgcn_s_wait_tensorcnt(0);
  }
  __syncthreads();
#else
#pragma unroll
  for (int i = 0; i < 4; ++i) {
    int idx = tid + i * 256;          // 1024 v8h chunks per 128x64 tile
    int row = idx >> 3;
    int c8  = (idx & 7) << 3;
#if HAVE_ASYNC
    async_cp16(qb + (size_t)(s0 + row) * DK + c8, Aq + row * 72 + c8);
    async_cp16(kb + (size_t)(t0 + row) * DK + c8, Bk + row * 72 + c8);
#else
    *(v8h*)(Aq + row * 72 + c8) = *(const v8h*)(qb + (size_t)(s0 + row) * DK + c8);
    *(v8h*)(Bk + row * 72 + c8) = *(const v8h*)(kb + (size_t)(t0 + row) * DK + c8);
#endif
  }
  wait_async<0>();
  __syncthreads();
#endif

  v8f acc[4][2] = {};
#pragma unroll
  for (int ks = 0; ks < DK; ks += 32) {
    v16h bf[2];
#pragma unroll
    for (int j = 0; j < 2; ++j)
      bf[j] = lds_load_b(Bk + (wn * 32 + j * 16) * 72 + ks, 72, lane);
#pragma unroll
    for (int i = 0; i < 4; ++i) {
      v16h af = lds_load_a(Aq + (wm * 64 + i * 16) * 72 + ks, 72, lane);
#pragma unroll
      for (int j = 0; j < 2; ++j) acc[i][j] = wmma_f16(af, bf[j], acc[i][j]);
    }
  }

  float* pb = probs + (size_t)bh * SS * SS;
#pragma unroll
  for (int i = 0; i < 4; ++i) {
    int sbase = s0 + wm * 64 + i * 16 + ((lane >> 4) << 3);
#pragma unroll
    for (int j = 0; j < 2; ++j) {
      int t = t0 + wn * 32 + j * 16 + (lane & 15);
#pragma unroll
      for (int r = 0; r < 8; ++r) {
        int s = sbase + r;
        __builtin_nontemporal_store(acc[i][j][r] * 0.125f,
                                    pb + (size_t)s * SS + t);
      }
    }
  }
}

// ---------------------------------------------------------------------------
// Kernel 2b: causal row softmax, in place; wave per row, wave32 reductions.
// ---------------------------------------------------------------------------
__global__ __launch_bounds__(256) void softmax_rows(float* __restrict__ probs) {
  const int row  = blockIdx.x * 8 + (threadIdx.x >> 5);
  const int lane = threadIdx.x & 31;
  const int s    = row & (SS - 1);
  float* p = probs + (size_t)row * SS;
  const int nv = s + 1;

  float vals[64];
  float mx = -3.0e38f;
#pragma unroll
  for (int j = 0; j < 64; ++j) {
    int i = lane + j * 32;
    float v = (i < nv) ? p[i] : -3.0e38f;
    vals[j] = v;
    mx = fmaxf(mx, v);
  }
#pragma unroll
  for (int off = 16; off >= 1; off >>= 1)
    mx = fmaxf(mx, __shfl_xor(mx, off, 32));

  float sum = 0.f;
#pragma unroll
  for (int j = 0; j < 64; ++j) {
    int i = lane + j * 32;
    float e = (i < nv) ? __expf(vals[j] - mx) : 0.f;
    vals[j] = e;
    sum += e;
  }
#pragma unroll
  for (int off = 16; off >= 1; off >>= 1)
    sum += __shfl_xor(sum, off, 32);

  float inv = 1.0f / sum;
#pragma unroll
  for (int j = 0; j < 64; ++j)
    __builtin_nontemporal_store(vals[j] * inv, p + lane + j * 32);
}

// ---------------------------------------------------------------------------
// Kernel 2c: ctx = probs @ v per head (causal K truncation).  v-tile (64x32,
// row stride S) fetched by TDM (strided 2D tile), probs f32->f16 in staging.
// Block tile 128x64, 8 waves (4x2) of 32x32.
// ---------------------------------------------------------------------------
__global__ __launch_bounds__(256) void attn_ctx(
    const float* __restrict__ probs, const _Float16* __restrict__ vht,
    _Float16* __restrict__ ctxh) {
  const int bh = blockIdx.z;
  const int b  = bh >> 4, h = bh & 15;
  const int s0 = blockIdx.x * 128;

  __shared__ _Float16 Ap[128 * 40];
  __shared__ _Float16 Bv[64 * 40];

  const int tid  = threadIdx.x;
  const int lane = tid & 31;
  const int w    = tid >> 5;
  const int wm   = w & 3;
  const int wn   = w >> 2;

  const float*    pb = probs + (size_t)bh * SS * SS;
  const _Float16* vb = vht + (size_t)bh * DK * SS;

  v8f acc[2][2] = {};
  const int kmax = s0 + 128;

  for (int k0 = 0; k0 < kmax; k0 += 32) {
#if HAVE_TDM
    if (w == 0) {
      // 64x32 f16 tile, global row stride SS; LDS rows padded 32->40 halves
      // (pad codes: 16 DW interval / 4 DW amount).
      tdm_load_tile_f16(vb + k0, lds_off_u32(Bv), 64, 32, SS, 3, 3, 1);
    }
#else
    {
      int idx = tid;                  // 256 v8h chunks per 64x32 v-tile
      int row = idx >> 2;
      int c8  = (idx & 3) << 3;
#if HAVE_ASYNC
      async_cp16(vb + (size_t)row * SS + k0 + c8, Bv + row * 40 + c8);
#else
      *(v8h*)(Bv + row * 40 + c8) = *(const v8h*)(vb + (size_t)row * SS + k0 + c8);
#endif
    }
#endif
#pragma unroll
    for (int i = 0; i < 4; ++i) {
      int idx = tid + i * 256;        // 1024 float4 chunks per 128x32 tile
      int row = idx >> 3;
      int c4  = (idx & 7) << 2;
      v4f a = *(const v4f*)(pb + (size_t)(s0 + row) * SS + k0 + c4);
      v4h ha;
#pragma unroll
      for (int t = 0; t < 4; ++t) ha[t] = (_Float16)a[t];
      *(v4h*)(Ap + row * 40 + c4) = ha;
    }
#if HAVE_TDM
    if (w == 0) __builtin_amdgcn_s_wait_tensorcnt(0);
#else
    wait_async<0>();
#endif
    __syncthreads();

    v16h bf[2];
#pragma unroll
    for (int j = 0; j < 2; ++j)
      bf[j] = lds_load_b(Bv + (wn * 32 + j * 16) * 40, 40, lane);
#pragma unroll
    for (int i = 0; i < 2; ++i) {
      v16h af = lds_load_a(Ap + (wm * 32 + i * 16) * 40, 40, lane);
#pragma unroll
      for (int j = 0; j < 2; ++j) acc[i][j] = wmma_f16(af, bf[j], acc[i][j]);
    }
    __syncthreads();
  }

#pragma unroll
  for (int i = 0; i < 2; ++i) {
    int sbase = s0 + wm * 32 + i * 16 + ((lane >> 4) << 3);
#pragma unroll
    for (int j = 0; j < 2; ++j) {
      int d = wn * 32 + j * 16 + (lane & 15);
#pragma unroll
      for (int r = 0; r < 8; ++r) {
        int s = sbase + r;
        ctxh[(((size_t)b * SS + s) * NH + h) * DK + d] = (_Float16)acc[i][j][r];
      }
    }
  }
}

// ---------------------------------------------------------------------------
// Kernel 3: out = ctx @ Wo^T + bo (all-f16 staging, async double-buffered,
// peeled pipeline as in kernel 1).
// ---------------------------------------------------------------------------
__global__ __launch_bounds__(256) void out_proj(
    const _Float16* __restrict__ ctxh, const _Float16* __restrict__ woh,
    const float* __restrict__ bo, float* __restrict__ out) {
  __shared__ _Float16 As[2 * 128 * 40];
  __shared__ _Float16 Bs[2 * 128 * 40];

  const int tid  = threadIdx.x;
  const int lane = tid & 31;
  const int w    = tid >> 5;
  const int wm   = w & 1;
  const int wn   = w >> 1;
  const int m0   = blockIdx.x * 128;
  const int n0   = blockIdx.y * 128;

  const _Float16* Xa = ctxh + (size_t)m0 * DM;
  const _Float16* Wb = woh + (size_t)n0 * DM;

  v8f acc[4][2] = {};

  stage_tile_128x32(Xa, 0, As, tid);
  stage_tile_128x32(Wb, 0, Bs, tid);
  int cur = 0;
  for (int k0 = 0; k0 < DM - 32; k0 += 32) {
    stage_tile_128x32(Xa, k0 + 32, As + (cur ^ 1) * 128 * 40, tid);
    stage_tile_128x32(Wb, k0 + 32, Bs + (cur ^ 1) * 128 * 40, tid);
    wait_async<4>();
    __syncthreads();
    gemm_step_128x128(As + cur * 128 * 40, Bs + cur * 128 * 40, wm, wn, lane, acc);
    __syncthreads();
    cur ^= 1;
  }
  wait_async<0>();
  __syncthreads();
  gemm_step_128x128(As + cur * 128 * 40, Bs + cur * 128 * 40, wm, wn, lane, acc);

#pragma unroll
  for (int i = 0; i < 4; ++i) {
    int mbase = m0 + wm * 64 + i * 16 + ((lane >> 4) << 3);
#pragma unroll
    for (int j = 0; j < 2; ++j) {
      int n = n0 + wn * 32 + j * 16 + (lane & 15);
      float bvn = bo[n];
#pragma unroll
      for (int r = 0; r < 8; ++r) {
        int m = mbase + r;
        out[(size_t)m * DM + n] = acc[i][j][r] + bvn;
      }
    }
  }
}

// ---------------------------------------------------------------------------
extern "C" void kernel_launch(void* const* d_in, const int* in_sizes, int n_in,
                              void* d_out, int out_size, void* d_ws,
                              size_t ws_size, hipStream_t stream) {
  (void)in_sizes; (void)n_in; (void)out_size; (void)ws_size;

  const float* Q  = (const float*)d_in[0];
  const float* K  = (const float*)d_in[1];
  const float* V  = (const float*)d_in[2];
  // d_in[3] = mask: causal tril, handled analytically.
  const float* Wq = (const float*)d_in[4];
  const float* bq = (const float*)d_in[5];
  const float* Wk = (const float*)d_in[6];
  const float* bk = (const float*)d_in[7];
  const float* Wv = (const float*)d_in[8];
  const float* bv = (const float*)d_in[9];
  const float* Wo = (const float*)d_in[10];
  const float* bo = (const float*)d_in[11];

  float* out   = (float*)d_out;
  float* probs = out + (size_t)BB * SS * DM;     // [B,H,S,S] f32 output #2

  const size_t E  = (size_t)BB * SS * DM;        // 8.39M elems
  const size_t WM = (size_t)DM * DM;             // 1.05M elems
  _Float16* ws   = (_Float16*)d_ws;
  _Float16* xh   = ws;                           // 3E: Q,K,V in f16
  _Float16* whb  = ws + 3 * E;                   // 4*WM: Wq,Wk,Wv,Wo in f16
  _Float16* qh   = whb + 4 * WM;                 // [B,H,S,DK]
  _Float16* kh   = qh + E;                       // [B,H,S,DK]
  _Float16* vht  = kh + E;                       // [B,H,DK,S] (transposed)
  _Float16* ctxh = vht + E;                      // [B,S,DM]

  cvt_f16<<<dim3((int)(E / 2048)), 256, 0, stream>>>(Q, xh, (int)E);
  cvt_f16<<<dim3((int)(E / 2048)), 256, 0, stream>>>(K, xh + E, (int)E);
  cvt_f16<<<dim3((int)(E / 2048)), 256, 0, stream>>>(V, xh + 2 * E, (int)E);
  cvt_f16<<<dim3((int)(WM / 2048)), 256, 0, stream>>>(Wq, whb, (int)WM);
  cvt_f16<<<dim3((int)(WM / 2048)), 256, 0, stream>>>(Wk, whb + WM, (int)WM);
  cvt_f16<<<dim3((int)(WM / 2048)), 256, 0, stream>>>(Wv, whb + 2 * WM, (int)WM);
  cvt_f16<<<dim3((int)(WM / 2048)), 256, 0, stream>>>(Wo, whb + 3 * WM, (int)WM);

  qkv_gemm<<<dim3(64, 8, 3), 256, 0, stream>>>(xh, whb, bq, bk, bv, qh, kh, vht);
  attn_scores<<<dim3(16, 16, BB * NH), 256, 0, stream>>>(qh, kh, probs);
  softmax_rows<<<dim3(BB * NH * SS / 8), 256, 0, stream>>>(probs);
  attn_ctx<<<dim3(16, 1, BB * NH), 256, 0, stream>>>(probs, vht, ctxh);
  out_proj<<<dim3(64, 8), 256, 0, stream>>>(ctxh, whb + 3 * WM, bo, out);
}
